// GatedDeltaNet_55611236549115
// MI455X (gfx1250) — matrix operand order repaired and checked by reading, not compiled, not run on hardware
//
#include <hip/hip_runtime.h>
#include <hip/hip_bf16.h>

// ---------- types ----------
typedef __attribute__((ext_vector_type(16))) __bf16 v16bf;
typedef __attribute__((ext_vector_type(8)))  __bf16 v8bf;
typedef __attribute__((ext_vector_type(8)))  float  v8f;
typedef __attribute__((ext_vector_type(4)))  float  f4_t;

#define HIDDEN_C   2048
#define NUM_V_C    16
#define NUM_K_C    8
#define HK_C       128
#define HV_C       128
#define KEY_DIM_C  1024
#define VALUE_DIM_C 2048
#define QKVZ_COLS  6144   // 2*KEY_DIM + 2*VALUE_DIM
#define CONV_DIM_C 4096

static __device__ __forceinline__ __bf16 to_bf16(float f) {
    unsigned u = __builtin_bit_cast(unsigned, f);
    u = u + 0x7FFFu + ((u >> 16) & 1u);          // round-to-nearest-even
    unsigned short h = (unsigned short)(u >> 16);
    return __builtin_bit_cast(__bf16, h);
}

// ---------- f32 -> bf16 conversion ----------
__global__ void f32_to_bf16_kern(const float* __restrict__ in, __bf16* __restrict__ out, size_t n) {
    size_t i = (size_t)blockIdx.x * blockDim.x + threadIdx.x;
    size_t stride = (size_t)gridDim.x * blockDim.x;
    for (; i < n; i += stride) out[i] = to_bf16(in[i]);
}

// ---------- WMMA fragment helpers ----------
// A fragment (16x32): lane l holds row M=l&15; K=(l>>4)*16+0..15
// (base pre-offset by lhalf*16)
static __device__ __forceinline__ void load_a4(const __bf16* __restrict__ base, int K,
                                               int k0, v16bf (&a)[4]) {
    #pragma unroll
    for (int mt = 0; mt < 4; ++mt) {
        const __bf16* p = base + (size_t)mt * 16 * K + k0;
        v8bf lo = *reinterpret_cast<const v8bf*>(p);
        v8bf hi = *reinterpret_cast<const v8bf*>(p + 8);
        a[mt] = __builtin_shufflevector(lo, hi,
            0,1,2,3,4,5,6,7,8,9,10,11,12,13,14,15);
    }
}
// B fragment (32x16): lane l holds col N=l&15; K=(l>>4)*16+0..15 (base pre-offset by lhalf*16)
static __device__ __forceinline__ void load_b4(const __bf16* __restrict__ base, int K,
                                               int k0, v16bf (&b)[4]) {
    #pragma unroll
    for (int nt = 0; nt < 4; ++nt)
        b[nt] = *reinterpret_cast<const v16bf*>(base + (size_t)nt * 16 * K + k0);
}
static __device__ __forceinline__ void wmma16(const v16bf (&a)[4], const v16bf (&b)[4],
                                              v8f (&acc)[4][4]) {
    #pragma unroll
    for (int mt = 0; mt < 4; ++mt)
        #pragma unroll
        for (int nt = 0; nt < 4; ++nt)
            acc[mt][nt] = __builtin_amdgcn_wmma_f32_16x16x32_bf16(
                false, a[mt], false, b[nt], (short)0, acc[mt][nt], false, false);
}

// ---------- bf16 WMMA GEMM: C[M,N] = A[M,K] * B[N,K]^T ----------
// block = 256 thr (8 waves), block tile 128x256, wave tile 64x64 (4x4 WMMA 16x16x32)
// software-pipelined: ping-pong fragment buffers overlap L2 latency with WMMAs.
// A,B row-major bf16, K contiguous. M%128==0, N%256==0, K%64==0.
__global__ __launch_bounds__(256) void gemm_bf16_wmma(
    const __bf16* __restrict__ A, const __bf16* __restrict__ B,
    float* __restrict__ C, int M, int N, int K)
{
    const int lane  = threadIdx.x & 31;
    const int wave  = threadIdx.x >> 5;
    const int lhalf = lane >> 4;      // 0 or 1
    const int l16   = lane & 15;
    const int waveM = blockIdx.y * 128 + (wave >> 2) * 64;
    const int waveN = blockIdx.x * 256 + (wave & 3) * 64;

    const __bf16* aBase = A + (size_t)(waveM + l16) * K + lhalf * 16;
    const __bf16* bBase = B + (size_t)(waveN + l16) * K + lhalf * 16;

    v8f acc[4][4] = {};
    v16bf A0[4], B0[4], A1[4], B1[4];

    load_a4(aBase, K, 0, A0);
    load_b4(bBase, K, 0, B0);
    int k0 = 0;
    for (;;) {
        // even stage: prefetch k0+32 into buffer 1, compute with buffer 0
        if (k0 + 32 < K) { load_a4(aBase, K, k0 + 32, A1); load_b4(bBase, K, k0 + 32, B1); }
        wmma16(A0, B0, acc);
        k0 += 32; if (k0 >= K) break;
        // odd stage: prefetch into buffer 0, compute with buffer 1
        if (k0 + 32 < K) { load_a4(aBase, K, k0 + 32, A0); load_b4(bBase, K, k0 + 32, B0); }
        wmma16(A1, B1, acc);
        k0 += 32; if (k0 >= K) break;
    }

    // C/D layout: lane<16 -> VGPR r = (M=r, N=lane); lane>=16 -> (M=8+r, N=lane-16)
    #pragma unroll
    for (int mt = 0; mt < 4; ++mt)
        #pragma unroll
        for (int nt = 0; nt < 4; ++nt)
            #pragma unroll
            for (int r = 0; r < 8; ++r) {
                int m = waveM + mt * 16 + lhalf * 8 + r;
                int n = waveN + nt * 16 + l16;
                C[(size_t)m * N + n] = acc[mt][nt][r];
            }
}

// ---------- ba projection: ba[t,32] = hidden[t,:] . w_ba[j,:] ----------
__global__ __launch_bounds__(256) void ba_gemm_kern(
    const float* __restrict__ hidden, const float* __restrict__ wba, float* __restrict__ ba)
{
    int warp = (blockIdx.x * blockDim.x + threadIdx.x) >> 5;   // [0, T*32)
    int lane = threadIdx.x & 31;
    int t = warp >> 5, j = warp & 31;
    const float* hp = hidden + (size_t)t * HIDDEN_C;
    const float* wp = wba + (size_t)j * HIDDEN_C;
    float s = 0.f;
    for (int k = lane; k < HIDDEN_C; k += 32) s += hp[k] * wp[k];
    #pragma unroll
    for (int m = 16; m >= 1; m >>= 1) s += __shfl_xor(s, m);
    if (lane == 0) ba[t * 32 + j] = s;
}

// ---------- reorder + depthwise causal conv1d (K=4) + silu ----------
// mixed channel c: [0,1024) q  [1024,2048) k  [2048,4096) v
__global__ __launch_bounds__(256) void conv_silu_kern(
    const float* __restrict__ qkvz, const float* __restrict__ convw,
    float* __restrict__ qbuf, float* __restrict__ kbuf, float* __restrict__ vbuf)
{
    int idx = blockIdx.x * blockDim.x + threadIdx.x;   // [0, T*4096)
    int c = idx & 4095;
    int t = idx >> 12;
    int col; float* dst; int di;
    if (c < 1024) {
        int kh = c >> 7, wi = c & 127;
        col = kh * 768 + wi; dst = qbuf; di = t * 1024 + c;
    } else if (c < 2048) {
        int cc = c - 1024; int kh = cc >> 7, wi = cc & 127;
        col = kh * 768 + 128 + wi; dst = kbuf; di = t * 1024 + cc;
    } else {
        int cv = c - 2048; int vh = cv >> 7, wi = cv & 127;
        col = (vh >> 1) * 768 + 256 + (vh & 1) * 128 + wi; dst = vbuf; di = t * 2048 + cv;
    }
    const float* w = convw + c * 4;
    float acc = 0.f;
    #pragma unroll
    for (int j = 0; j < 4; ++j) {
        int tt = t - 3 + j;
        if (tt >= 0) acc += qkvz[(size_t)tt * QKVZ_COLS + col] * w[j];
    }
    acc = acc / (1.f + __expf(-acc));   // silu
    dst[di] = acc;
}

// ---------- L2 norm over rows of 128 (in place), with output scale ----------
__global__ __launch_bounds__(256) void l2norm_rows_kern(float* __restrict__ buf, int rows, float outscale) {
    int warp = (blockIdx.x * blockDim.x + threadIdx.x) >> 5;
    int lane = threadIdx.x & 31;
    if (warp >= rows) return;
    float* p = buf + (size_t)warp * 128;
    float v[4]; float ss = 0.f;
    #pragma unroll
    for (int i = 0; i < 4; ++i) { v[i] = p[lane + 32 * i]; ss += v[i] * v[i]; }
    #pragma unroll
    for (int m = 16; m >= 1; m >>= 1) ss += __shfl_xor(ss, m);
    float inv = rsqrtf(ss + 1e-6f) * outscale;
    #pragma unroll
    for (int i = 0; i < 4; ++i) p[lane + 32 * i] = v[i] * inv;
}

// ---------- gates: beta = sigmoid(b), g = -exp(A_log)*softplus(a+dt_bias) ----------
__global__ void gates_kern(const float* __restrict__ ba, const float* __restrict__ dt_bias,
                           const float* __restrict__ A_log,
                           float* __restrict__ beta, float* __restrict__ g)
{
    int i = blockIdx.x * blockDim.x + threadIdx.x;   // [0, T*16)
    int t = i >> 4, vh = i & 15;
    int kh = vh >> 1, sub = vh & 1;
    float b = ba[t * 32 + kh * 4 + sub];
    float a = ba[t * 32 + kh * 4 + 2 + sub];
    beta[i] = 1.f / (1.f + __expf(-b));
    float x = a + dt_bias[vh];
    float sp = (x > 20.f) ? x : log1pf(__expf(x));
    g[i] = -__expf(A_log[vh]) * sp;
}

// ---------- gated delta-rule recurrence ----------
// grid: 64 blocks = 16 heads x 4 v-chunks; 256 thr; lane owns S[kbase..kbase+15][v]
// reductions over k via __shfl_xor (8 lanes, stride 4) -> zero barriers in T loop
__global__ __launch_bounds__(256) void delta_recurrence_kern(
    const float* __restrict__ qn, const float* __restrict__ kn,
    const float* __restrict__ vbuf, const float* __restrict__ g,
    const float* __restrict__ beta, float* __restrict__ obuf, int T)
{
    const int h = blockIdx.x >> 2;
    const int vchunk = blockIdx.x & 3;
    const int kh = h >> 1;
    const int wave = threadIdx.x >> 5;
    const int lane = threadIdx.x & 31;
    const int v = vchunk * 32 + wave * 4 + (lane & 3);
    const int kbase = (lane >> 2) * 16;

    float S[16];
    #pragma unroll
    for (int i = 0; i < 16; ++i) S[i] = 0.f;

    for (int t = 0; t < T; ++t) {
        const float gt = __expf(g[t * 16 + h]);
        const float bt = beta[t * 16 + h];
        const float vt = vbuf[((size_t)t * 16 + h) * 128 + v];
        const f4_t* kp = reinterpret_cast<const f4_t*>(kn + ((size_t)t * 8 + kh) * 128 + kbase);
        const f4_t* qp = reinterpret_cast<const f4_t*>(qn + ((size_t)t * 8 + kh) * 128 + kbase);
        f4_t kf[4] = {kp[0], kp[1], kp[2], kp[3]};
        f4_t qf[4] = {qp[0], qp[1], qp[2], qp[3]};

        float kv = 0.f;
        #pragma unroll
        for (int i = 0; i < 16; ++i) {
            S[i] *= gt;
            kv += kf[i >> 2][i & 3] * S[i];
        }
        kv += __shfl_xor(kv, 4); kv += __shfl_xor(kv, 8); kv += __shfl_xor(kv, 16);
        const float delta = (vt - kv) * bt;
        float o = 0.f;
        #pragma unroll
        for (int i = 0; i < 16; ++i) {
            S[i] += kf[i >> 2][i & 3] * delta;
            o += qf[i >> 2][i & 3] * S[i];     // q already carries HK^-0.5
        }
        o += __shfl_xor(o, 4); o += __shfl_xor(o, 8); o += __shfl_xor(o, 16);
        if (lane < 4) obuf[((size_t)t * 16 + h) * 128 + v] = o;
    }
}

// ---------- gated RMSNorm: x = o*silu(z); xn = x*rsqrt(mean(x^2)+eps)*w -> bf16 ----------
__global__ __launch_bounds__(256) void gated_rmsnorm_kern(
    const float* __restrict__ obuf, const float* __restrict__ qkvz,
    const float* __restrict__ nw, __bf16* __restrict__ xn)
{
    int warp = (blockIdx.x * blockDim.x + threadIdx.x) >> 5;   // [0, T*16)
    int lane = threadIdx.x & 31;
    int t = warp >> 4, vh = warp & 15;
    int kh = vh >> 1, sub = vh & 1;
    size_t obase = ((size_t)t * 16 + vh) * 128;
    size_t zbase = (size_t)t * QKVZ_COLS + kh * 768 + 512 + sub * 128;
    float x[4]; float ss = 0.f;
    #pragma unroll
    for (int i = 0; i < 4; ++i) {
        int wi = lane + 32 * i;
        float z = qkvz[zbase + wi];
        float o = obuf[obase + wi];
        x[i] = o * (z / (1.f + __expf(-z)));
        ss += x[i] * x[i];
    }
    #pragma unroll
    for (int m = 16; m >= 1; m >>= 1) ss += __shfl_xor(ss, m);
    float inv = rsqrtf(ss * (1.f / 128.f) + 1e-6f);
    #pragma unroll
    for (int i = 0; i < 4; ++i) {
        int wi = lane + 32 * i;
        xn[(size_t)t * VALUE_DIM_C + vh * 128 + wi] = to_bf16(x[i] * inv * nw[wi]);
    }
}

// =======================================================================
extern "C" void kernel_launch(void* const* d_in, const int* in_sizes, int n_in,
                              void* d_out, int out_size, void* d_ws, size_t ws_size,
                              hipStream_t stream) {
    (void)n_in; (void)out_size; (void)ws_size;
    const float* hidden  = (const float*)d_in[0];
    const float* w_qkvz  = (const float*)d_in[1];
    const float* w_ba    = (const float*)d_in[2];
    const float* conv_w  = (const float*)d_in[3];
    const float* dt_bias = (const float*)d_in[4];
    const float* A_log   = (const float*)d_in[5];
    const float* norm_w  = (const float*)d_in[6];
    const float* w_out   = (const float*)d_in[7];
    float* out = (float*)d_out;

    const int T = in_sizes[0] / HIDDEN_C;   // 1024

    char* ws = (char*)d_ws;
    size_t off = 0;
    auto alloc = [&](size_t bytes) -> void* {
        void* p = ws + off;
        off += (bytes + 255) & ~(size_t)255;
        return p;
    };
    __bf16* hbf    = (__bf16*)alloc((size_t)T * HIDDEN_C * 2);
    __bf16* wqkbf  = (__bf16*)alloc((size_t)QKVZ_COLS * HIDDEN_C * 2);
    __bf16* woutbf = (__bf16*)alloc((size_t)HIDDEN_C * VALUE_DIM_C * 2);
    float*  qkvz   = (float*) alloc((size_t)T * QKVZ_COLS * 4);
    float*  ba     = (float*) alloc((size_t)T * 32 * 4);
    float*  qn     = (float*) alloc((size_t)T * KEY_DIM_C * 4);
    float*  kn     = (float*) alloc((size_t)T * KEY_DIM_C * 4);
    float*  vb     = (float*) alloc((size_t)T * VALUE_DIM_C * 4);
    float*  gbuf   = (float*) alloc((size_t)T * NUM_V_C * 4);
    float*  bbuf   = (float*) alloc((size_t)T * NUM_V_C * 4);
    float*  ob     = (float*) alloc((size_t)T * VALUE_DIM_C * 4);
    __bf16* xnbf   = (__bf16*)alloc((size_t)T * VALUE_DIM_C * 2);

    // 1) f32 -> bf16 conversions (acts + both GEMM weights)
    {
        size_t n1 = (size_t)T * HIDDEN_C;
        size_t n2 = (size_t)QKVZ_COLS * HIDDEN_C;
        size_t n3 = (size_t)HIDDEN_C * VALUE_DIM_C;
        f32_to_bf16_kern<<<(unsigned)((n1 + 255) / 256), 256, 0, stream>>>(hidden, hbf, n1);
        f32_to_bf16_kern<<<(unsigned)((n2 + 255) / 256), 256, 0, stream>>>(w_qkvz, wqkbf, n2);
        f32_to_bf16_kern<<<(unsigned)((n3 + 255) / 256), 256, 0, stream>>>(w_out, woutbf, n3);
    }
    // 2) qkvz projection GEMM (WMMA bf16): [T,2048] x [6144,2048]^T
    gemm_bf16_wmma<<<dim3(QKVZ_COLS / 256, T / 128), 256, 0, stream>>>(
        hbf, wqkbf, qkvz, T, QKVZ_COLS, HIDDEN_C);
    // 3) ba projection
    ba_gemm_kern<<<(T * 32) / 8, 256, 0, stream>>>(hidden, w_ba, ba);
    // 4) reorder + depthwise causal conv + silu
    conv_silu_kern<<<(T * CONV_DIM_C) / 256, 256, 0, stream>>>(qkvz, conv_w, qn, kn, vb);
    // 5) L2 norm q (fold in HK^-0.5) and k
    l2norm_rows_kern<<<(T * NUM_K_C * 32) / 256, 256, 0, stream>>>(qn, T * NUM_K_C, 0.08838834764831845f);
    l2norm_rows_kern<<<(T * NUM_K_C * 32) / 256, 256, 0, stream>>>(kn, T * NUM_K_C, 1.0f);
    // 6) gates
    gates_kern<<<(T * NUM_V_C) / 256, 256, 0, stream>>>(ba, dt_bias, A_log, bbuf, gbuf);
    // 7) gated delta-rule recurrence (serial in T, parallel over head x v-chunk)
    delta_recurrence_kern<<<NUM_V_C * 4, 256, 0, stream>>>(qn, kn, vb, gbuf, bbuf, ob, T);
    // 8) gated RMSNorm -> bf16 activations
    gated_rmsnorm_kern<<<(T * NUM_V_C * 32) / 256, 256, 0, stream>>>(ob, qkvz, norm_w, xnbf);
    // 9) output projection GEMM (WMMA bf16): [T,2048] x [2048,2048]^T -> d_out
    gemm_bf16_wmma<<<dim3(HIDDEN_C / 256, T / 128), 256, 0, stream>>>(
        xnbf, woutbf, out, T, HIDDEN_C, VALUE_DIM_C);
}
